// MyBlock_46935402611303
// MI455X (gfx1250) — matrix operand order repaired and checked by reading, not compiled, not run on hardware
//
#include <hip/hip_runtime.h>
#include <hip/hip_bf16.h>

// ---------------------------------------------------------------------------
// Mamba block for MI455X (gfx1250, wave32).
// GEMMs: v_wmma_f32_16x16x32_bf16 (bf16 in, f32 accum). Elementwise: fp32.
// All GEMM shapes padded so M%128==0 and N%64==0 -> branch-free inner loops.
// ---------------------------------------------------------------------------

#define T_TOK   2048   // B*L
#define SEQ_L   1024
#define DMODEL  1024
#define DINNER  2048
#define DTRANK  64
#define DSTATE  16
#define DFFN    4096
#define XZCOLS  4096   // 2*DINNER
#define XDBLP   128    // padded DTRANK + 2*DSTATE (96 -> 128)

typedef __bf16 bf16;
typedef __attribute__((ext_vector_type(16))) __bf16 v16bf;
typedef __attribute__((ext_vector_type(8)))  __bf16 v8bf;
typedef __attribute__((ext_vector_type(8)))  float  v8f;

__device__ __forceinline__ bf16 f2bf(float x) {
    union { float f; unsigned u; } v; v.f = x;
    unsigned r = v.u + 0x7FFFu + ((v.u >> 16) & 1u);   // round-to-nearest-even
    union { unsigned short s; bf16 b; } o;
    o.s = (unsigned short)(r >> 16);
    return o.b;
}

__device__ __forceinline__ float silu_f(float x) {
    return x / (1.f + __expf(-x));
}

// ---------------------------------------------------------------------------
// f32 -> bf16 conversion
// ---------------------------------------------------------------------------
__global__ void f32_to_bf16_kernel(const float* __restrict__ src,
                                   bf16* __restrict__ dst, int n) {
    for (int i = blockIdx.x * blockDim.x + threadIdx.x; i < n;
         i += gridDim.x * blockDim.x)
        dst[i] = f2bf(src[i]);
}

// f32 [rows,cols] -> bf16 [padRows,cols], rows >= actual are zero-filled
__global__ void f32_to_bf16_pad_kernel(const float* __restrict__ src,
                                       bf16* __restrict__ dst,
                                       int rows, int cols, int padRows) {
    const int total = padRows * cols;
    for (int i = blockIdx.x * blockDim.x + threadIdx.x; i < total;
         i += gridDim.x * blockDim.x) {
        const int r = i / cols;
        const int c = i - r * cols;
        dst[i] = (r < rows) ? f2bf(src[(size_t)r * cols + c]) : f2bf(0.f);
    }
}

// ---------------------------------------------------------------------------
// LayerNorm over last dim (D=1024), bf16 output for the following GEMM
// ---------------------------------------------------------------------------
__global__ __launch_bounds__(256)
void layernorm_bf16_kernel(const float* __restrict__ x,
                           const float* __restrict__ w,
                           bf16* __restrict__ out) {
    const int D = DMODEL;
    const int row = blockIdx.x;
    const float* xr = x + (size_t)row * D;
    float s = 0.f, ss = 0.f;
    for (int i = threadIdx.x; i < D; i += 256) {
        float v = xr[i]; s += v; ss += v * v;
    }
    __shared__ float rs[256], rss[256];
    rs[threadIdx.x] = s; rss[threadIdx.x] = ss;
    __syncthreads();
    for (int off = 128; off > 0; off >>= 1) {
        if (threadIdx.x < off) {
            rs[threadIdx.x]  += rs[threadIdx.x + off];
            rss[threadIdx.x] += rss[threadIdx.x + off];
        }
        __syncthreads();
    }
    const float mu   = rs[0] * (1.f / D);
    const float var  = rss[0] * (1.f / D) - mu * mu;
    const float rinv = rsqrtf(var + 1e-5f);
    for (int i = threadIdx.x; i < D; i += 256)
        out[(size_t)row * D + i] = f2bf((xr[i] - mu) * rinv * w[i]);
}

// ---------------------------------------------------------------------------
// WMMA bf16 NT GEMM:  D[M,N] = A[M,K] * W[N,K]^T  (+ fused epilogue)
// Requires M%128==0, N%64==0, K%32==0 (shapes arranged so this always holds).
//   A fragment (16x32): lane l -> row (l&15); 16B chunks at K {0,8}/{16,24}
//   B fragment (32x16): lane l -> row (l&15) of W; 32B chunk at K (l>>4)*16
//   C/D:               lane l -> col (l&15); vgpr r -> row r + 8*(l>>4)
// block = 256 thr = 8 waves; block tile 128(M) x 64(N); wave -> 16x64 strip.
// n0 depends only on blockIdx -> zero divergent control flow in the kernel.
// ep: 0 none | 1 softplus(x + bias[n]) | 2 silu | 3 += res[m*ldc+n]
// ---------------------------------------------------------------------------
union FragA { v16bf v; struct { v8bf lo, hi; } p; };

__global__ __launch_bounds__(256)
void gemm_bf16_wmma(const bf16* __restrict__ A, int lda,
                    const bf16* __restrict__ W, int ldw,
                    float* __restrict__ Df, bf16* __restrict__ Dbf, int ldc,
                    int K, int ep, const float* __restrict__ epp) {
    const int lane = threadIdx.x & 31;
    const int wid  = threadIdx.x >> 5;            // 0..7
    const int m0   = blockIdx.y * 128 + wid * 16; // wave's M strip
    const int n0   = blockIdx.x * 64;             // block's N strip (scalar)

    const int row16 = lane & 15;
    const int khalfA = (lane >> 4) << 4;          // 0 / 16
    const int khalfB = (lane >> 4) << 4;          // 0 / 16

    const bf16* ap  = A + (size_t)(m0 + row16) * lda + khalfA;
    const bf16* bp0 = W + (size_t)(n0      + row16) * ldw + khalfB;
    const bf16* bp1 = W + (size_t)(n0 + 16 + row16) * ldw + khalfB;
    const bf16* bp2 = W + (size_t)(n0 + 32 + row16) * ldw + khalfB;
    const bf16* bp3 = W + (size_t)(n0 + 48 + row16) * ldw + khalfB;

    const v8f vzero = {0.f, 0.f, 0.f, 0.f, 0.f, 0.f, 0.f, 0.f};
    v8f acc[4];
#pragma unroll
    for (int j = 0; j < 4; ++j) acc[j] = vzero;

#pragma unroll 2
    for (int k = 0; k < K; k += 32) {
        FragA a;
        a.p.lo = *reinterpret_cast<const v8bf*>(ap + k);
        a.p.hi = *reinterpret_cast<const v8bf*>(ap + k + 8);
        __builtin_prefetch(ap + k + 32, 0, 1);          // global_prefetch_b8
        const v16bf b0 = *reinterpret_cast<const v16bf*>(bp0 + k);
        const v16bf b1 = *reinterpret_cast<const v16bf*>(bp1 + k);
        const v16bf b2 = *reinterpret_cast<const v16bf*>(bp2 + k);
        const v16bf b3 = *reinterpret_cast<const v16bf*>(bp3 + k);
        __builtin_prefetch(bp0 + k + 32, 0, 1);
        acc[0] = __builtin_amdgcn_wmma_f32_16x16x32_bf16(
            false, a.v, false, b0, (short)0, acc[0], false, false);
        acc[1] = __builtin_amdgcn_wmma_f32_16x16x32_bf16(
            false, a.v, false, b1, (short)0, acc[1], false, false);
        acc[2] = __builtin_amdgcn_wmma_f32_16x16x32_bf16(
            false, a.v, false, b2, (short)0, acc[2], false, false);
        acc[3] = __builtin_amdgcn_wmma_f32_16x16x32_bf16(
            false, a.v, false, b3, (short)0, acc[3], false, false);
    }

    const int col   = n0 + row16;
    const int rbase = m0 + ((lane >> 4) << 3);
#pragma unroll
    for (int j = 0; j < 4; ++j) {
        const int c = col + j * 16;
#pragma unroll
        for (int r = 0; r < 8; ++r) {
            float v = acc[j][r];
            const size_t idx = (size_t)(rbase + r) * ldc + c;
            if (ep == 1) {                         // softplus(x + bias[n])
                float t = v + epp[c];
                v = (t > 20.f) ? t : log1pf(__expf(t));
            } else if (ep == 2) {                  // silu
                v = silu_f(v);
            } else if (ep == 3) {                  // residual add
                v += epp[idx];
            }
            if (Df)  Df[idx]  = v;
            if (Dbf) Dbf[idx] = f2bf(v);
        }
    }
}

// ---------------------------------------------------------------------------
// Causal depthwise conv1d (width 4) + bias + SiLU.  xr = xz[:, :DINNER].
// ---------------------------------------------------------------------------
__global__ __launch_bounds__(256)
void conv_silu_kernel(const float* __restrict__ xz,
                      const float* __restrict__ cw,
                      const float* __restrict__ cb,
                      float* __restrict__ xc, bf16* __restrict__ xcb) {
    const int idx = blockIdx.x * blockDim.x + threadIdx.x;   // T_TOK*DINNER
    if (idx >= T_TOK * DINNER) return;
    const int d   = idx & (DINNER - 1);
    const int tok = idx >> 11;                    // /DINNER
    const int l   = tok & (SEQ_L - 1);
    float accv = cb[d];
#pragma unroll
    for (int i = 0; i < 4; ++i) {
        const int ls = l - 3 + i;
        if (ls >= 0)
            accv += cw[d * 4 + i] * xz[(size_t)(tok - 3 + i) * XZCOLS + d];
    }
    const float s = silu_f(accv);
    xc[idx]  = s;
    xcb[idx] = f2bf(s);
}

// ---------------------------------------------------------------------------
// Selective scan: one lane per (b,d) channel, 16 states in VGPRs.
// Stages B_t / C_t (16 floats each) through LDS per timestep.
// Output y = (scan + Dp*xc) * silu(z), stored bf16 for out_proj GEMM.
// ---------------------------------------------------------------------------
__global__ __launch_bounds__(256)
void scan_kernel(const float* __restrict__ dtp,
                 const float* __restrict__ xdbl,
                 const float* __restrict__ xc,
                 const float* __restrict__ xz,
                 const float* __restrict__ A_log,
                 const float* __restrict__ Dp,
                 bf16* __restrict__ ybf) {
    const int b = blockIdx.y;
    const int d = blockIdx.x * 256 + threadIdx.x;  // 0..DINNER-1
    float Ad[DSTATE], h[DSTATE];
#pragma unroll
    for (int n = 0; n < DSTATE; ++n) {
        Ad[n] = -__expf(A_log[(size_t)d * DSTATE + n]);
        h[n]  = 0.f;
    }
    const float Dd = Dp[d];
    __shared__ float sB[DSTATE], sC[DSTATE];

    for (int l = 0; l < SEQ_L; ++l) {
        const int tok = b * SEQ_L + l;
        if (threadIdx.x < 32) {
            const int n = threadIdx.x & 15;
            const float v = xdbl[(size_t)tok * XDBLP + DTRANK +
                                 (threadIdx.x < 16 ? n : DSTATE + n)];
            if (threadIdx.x < 16) sB[n] = v; else sC[n] = v;
        }
        __syncthreads();
        const float dtv = dtp[(size_t)tok * DINNER + d];
        const float xcv = xc [(size_t)tok * DINNER + d];
        const float zv  = xz [(size_t)tok * XZCOLS + DINNER + d];
        float y = 0.f;
#pragma unroll
        for (int n = 0; n < DSTATE; ++n) {
            h[n] = __expf(dtv * Ad[n]) * h[n] + dtv * sB[n] * xcv;
            y += h[n] * sC[n];
        }
        y = (y + Dd * xcv) * silu_f(zv);
        ybf[(size_t)tok * DINNER + d] = f2bf(y);
        __syncthreads();
    }
}

// ---------------------------------------------------------------------------
// Launch
// ---------------------------------------------------------------------------
extern "C" void kernel_launch(void* const* d_in, const int* in_sizes, int n_in,
                              void* d_out, int out_size, void* d_ws, size_t ws_size,
                              hipStream_t stream) {
    (void)in_sizes; (void)n_in; (void)out_size; (void)ws_size;
    const float* x          = (const float*)d_in[0];
    const float* ln1_w      = (const float*)d_in[1];
    const float* in_proj_w  = (const float*)d_in[2];
    const float* conv_w     = (const float*)d_in[3];
    const float* conv_b     = (const float*)d_in[4];
    const float* x_proj_w   = (const float*)d_in[5];
    const float* dt_proj_w  = (const float*)d_in[6];
    const float* dt_proj_b  = (const float*)d_in[7];
    const float* A_log      = (const float*)d_in[8];
    const float* Dp         = (const float*)d_in[9];
    const float* out_proj_w = (const float*)d_in[10];
    const float* ln2_w      = (const float*)d_in[11];
    const float* fc_w       = (const float*)d_in[12];
    const float* proj_w     = (const float*)d_in[13];
    float* out = (float*)d_out;

    char* p = (char*)d_ws;
    auto alloc = [&](size_t bytes) -> void* {
        void* r = (void*)p;
        p += (bytes + 255) & ~(size_t)255;
        return r;
    };
    bf16*  w_in  = (bf16*) alloc((size_t)XZCOLS * DMODEL * 2);
    bf16*  w_xp  = (bf16*) alloc((size_t)XDBLP  * DINNER * 2);  // 96->128 rows, padded
    bf16*  w_dt  = (bf16*) alloc((size_t)DINNER * DTRANK * 2);
    bf16*  w_out = (bf16*) alloc((size_t)DMODEL * DINNER * 2);
    bf16*  w_fc  = (bf16*) alloc((size_t)DFFN   * DMODEL * 2);
    bf16*  w_pr  = (bf16*) alloc((size_t)DMODEL * DFFN   * 2);
    bf16*  xn    = (bf16*) alloc((size_t)T_TOK * DMODEL * 2);
    float* xz    = (float*)alloc((size_t)T_TOK * XZCOLS * 4);
    float* xc    = (float*)alloc((size_t)T_TOK * DINNER * 4);
    bf16*  xcb   = (bf16*) alloc((size_t)T_TOK * DINNER * 2);
    float* xdbl  = (float*)alloc((size_t)T_TOK * XDBLP  * 4);   // stride 128
    bf16*  xdblb = (bf16*) alloc((size_t)T_TOK * XDBLP  * 2);
    float* dtb   = (float*)alloc((size_t)T_TOK * DINNER * 4);
    bf16*  ybf   = (bf16*) alloc((size_t)T_TOK * DINNER * 2);
    float* out1  = (float*)alloc((size_t)T_TOK * DMODEL * 4);
    bf16*  h2b   = (bf16*) alloc((size_t)T_TOK * DMODEL * 2);
    bf16*  h3b   = (bf16*) alloc((size_t)T_TOK * DFFN   * 2);

    auto cvt = [&](const float* s, bf16* dst, int n) {
        int g = (n + 255) / 256; if (g > 4096) g = 4096;
        f32_to_bf16_kernel<<<g, 256, 0, stream>>>(s, dst, n);
    };
    auto gemm = [&](const bf16* A, int lda, const bf16* W, int ldw,
                    float* Df, bf16* Dbf, int ldc, int M, int N, int K,
                    int ep, const float* epp) {
        dim3 grid(N / 64, M / 128);
        gemm_bf16_wmma<<<grid, 256, 0, stream>>>(A, lda, W, ldw, Df, Dbf, ldc,
                                                 K, ep, epp);
    };

    // weight conversions (f32 -> bf16)
    cvt(in_proj_w,  w_in,  XZCOLS * DMODEL);
    f32_to_bf16_pad_kernel<<<4096, 256, 0, stream>>>(x_proj_w, w_xp,
                                                     96, DINNER, XDBLP);
    cvt(dt_proj_w,  w_dt,  DINNER * DTRANK);
    cvt(out_proj_w, w_out, DMODEL * DINNER);
    cvt(fc_w,       w_fc,  DFFN   * DMODEL);
    cvt(proj_w,     w_pr,  DMODEL * DFFN);

    // --- Mamba mixer ---
    layernorm_bf16_kernel<<<T_TOK, 256, 0, stream>>>(x, ln1_w, xn);
    gemm(xn, DMODEL, w_in, DMODEL, xz, nullptr, XZCOLS,
         T_TOK, XZCOLS, DMODEL, 0, nullptr);                       // in_proj
    conv_silu_kernel<<<(T_TOK * DINNER) / 256, 256, 0, stream>>>(
        xz, conv_w, conv_b, xc, xcb);                              // conv + silu
    gemm(xcb, DINNER, w_xp, DINNER, xdbl, xdblb, XDBLP,
         T_TOK, XDBLP, DINNER, 0, nullptr);                        // x_proj (padded N)
    gemm(xdblb, XDBLP, w_dt, DTRANK, dtb, nullptr, DINNER,
         T_TOK, DINNER, DTRANK, 1, dt_proj_b);                     // dt_proj+softplus
    scan_kernel<<<dim3(DINNER / 256, 2), 256, 0, stream>>>(
        dtb, xdbl, xc, xz, A_log, Dp, ybf);                        // selective scan
    gemm(ybf, DINNER, w_out, DINNER, out1, nullptr, DMODEL,
         T_TOK, DMODEL, DINNER, 3, x);                             // out_proj + res

    // --- MLP ---
    layernorm_bf16_kernel<<<T_TOK, 256, 0, stream>>>(out1, ln2_w, h2b);
    gemm(h2b, DMODEL, w_fc, DMODEL, nullptr, h3b, DFFN,
         T_TOK, DFFN, DMODEL, 2, nullptr);                         // fc + silu
    gemm(h3b, DFFN, w_pr, DFFN, out, nullptr, DMODEL,
         T_TOK, DMODEL, DFFN, 3, out1);                            // proj + res
}
